// LSRecurrentScanDrivenGPT_51127290691991
// MI455X (gfx1250) — matrix operand order repaired and checked by reading, not compiled, not run on hardware
//
// MI455X (gfx1250) implementation of LSRecurrentScanDrivenGPT forward.
//
// Strategy (reasoned from MI455X specs, compile-only):
//  * All matmuls via v_wmma_f32_16x16x32_bf16 with an f32->(bf16 hi, bf16 lo)
//    split and a 3-WMMA product (hi*hi + hi*lo + lo*hi) => ~f32 accuracy at
//    bf16 matrix-engine rate. lm_head (211 GFLOP, 206MB weights) dominates;
//    bf16 WMMA keeps it near the 23.3 TB/s HBM roofline.
//  * Global->LDS staging via the Tensor Data Mover: wave 0 issues 2D-tile
//    tensor_load_to_lds DMAs (D# descriptor per ISA ch.8: 4B elements,
//    tile 32 x rows, tensor_dim0_stride = ld, pad 1 DWORD per 32 DWORDs so
//    LDS rows stride 33 words -> bank-conflict-free fragment reads),
//    double-buffered: chunk i+1 DMA overlaps chunk i WMMAs (TENSORcnt).
//    This toolchain exposes the 6-arg builtin (extra int32x8 group, zeroed).
//  * f32->bf16 hi/lo split happens in the fragment loader (VALU cvt
//    co-executes with XDL WMMA per ISA co-execution rules).
//  * Causal exponential scan = chunked linear recurrence (16 chunks x 128),
//    carry-combined per channel, fixup fused into rmsnorm(h).
//  * One generic tiled GEMM kernel: C = A[M,K] * B[N,K]^T, grid.z walks the
//    block-diagonal blocks, fused epilogues (SiLU, gain*SiLU, accumulate).

#include <hip/hip_runtime.h>
#include <math.h>

typedef __bf16 bf16;
typedef __attribute__((ext_vector_type(16))) __bf16        v16bf;
typedef __attribute__((ext_vector_type(8)))  float         v8f;
typedef __attribute__((ext_vector_type(4)))  unsigned int  v4u;
typedef __attribute__((ext_vector_type(8)))  int           v8i;
typedef __attribute__((ext_vector_type(4)))  int           v4i;

#define TT     2048
#define DD     1024
#define HH     1024
#define NBLK   16
#define VV     50304
#define RNK    64
#define NLAYER 4
#define CATW   3072   /* H + D + D */
#define NMEM   64
#define NFORG  960
#define SCHUNK 16     /* scan chunks */
#define SLEN   128    /* T / SCHUNK  */

#define BM   128      /* rows per workgroup  (4 wave-rows x 32) */
#define BN   64       /* cols per workgroup  (2 wave-cols x 32) */
#define KC   32       /* K chunk (one bf16 WMMA K)              */
#define LDSW (KC + 1) /* LDS row stride in f32 words (TDM pad)  */

// ---------------------------------------------------------------------------
// Tensor Data Mover: load a [rows x KC] f32 tile (row stride = strideElems)
// from global into LDS, packed with 1 DWORD pad per 32 DWORDs (row stride 33).
// D# bitfields per CDNA5 ISA ch.8 (group0: count/lds_addr/global_addr/type=2;
// group1: data_size=4B, pad_enable, pad_interval=32DW, pad_amount=1DW,
// tensor_dim0 = strideElems, tensor_dim1 = rows, tile_dim0 = KC,
// tile_dim1 = rows, tensor_dim0_stride = strideElems).
// Issued per-wave (EXEC ignored): call from ONE wave only. Tracked: TENSORcnt.
// This toolchain's builtin takes 6 args: (g0, g1, g2, g3, g_extra, cpol);
// groups beyond g1 are unused for a 2D tile -> zero-filled.
// ---------------------------------------------------------------------------
__device__ __forceinline__ void tdm_load_tile(const float* gsrc, float* lds_ptr,
                                              unsigned int rows,
                                              unsigned int strideElems) {
  unsigned long long ga = (unsigned long long)(uintptr_t)gsrc;
  unsigned int ldsOff   = (unsigned int)(uintptr_t)lds_ptr; // low 32 = LDS offset
  v4u g0;
  g0[0] = 1u;                                              // count = 1 (valid)
  g0[1] = ldsOff;                                          // lds_addr
  g0[2] = (unsigned int)ga;                                // global_addr[31:0]
  g0[3] = (unsigned int)((ga >> 32) & 0x01FFFFFFull) | (2u << 30); // [56:32]|type=2
  unsigned int dim0 = strideElems;                         // tensor width (elems)
  unsigned int dim1 = rows;                                // tensor rows
  v8i g1;
  g1[0] = (int)((2u << 16) | (1u << 20) | (4u << 22));     // 4B, pad_en, every 32DW, +1DW
  g1[1] = (int)((dim0 & 0xFFFFu) << 16);                   // tensor_dim0[15:0]
  g1[2] = (int)(((dim0 >> 16) & 0xFFFFu) | ((dim1 & 0xFFFFu) << 16));
  g1[3] = (int)(((dim1 >> 16) & 0xFFFFu) | ((unsigned)KC << 16)); // tile_dim0 = KC
  g1[4] = (int)(rows & 0xFFFFu);                           // tile_dim1; tile_dim2=0
  g1[5] = (int)dim0;                                       // tensor_dim0_stride[31:0]
  g1[6] = 0;                                               // stride[47:32], dim1_stride lo
  g1[7] = 0;
  v4i g2 = {0, 0, 0, 0};
  v4i g3 = {0, 0, 0, 0};
  v8i gx = {0, 0, 0, 0, 0, 0, 0, 0};
  __builtin_amdgcn_tensor_load_to_lds(g0, g1, g2, g3, gx, 0);
}

// ---------------------------------------------------------------------------
// Fragment loaders: read f32 from LDS (row stride LDSW) and split to bf16
// hi/lo in registers. Wave32 layouts per CDNA5 ISA 7.12.2:
// A 16x32: lane L -> row base+(L&15), K = kb..kb+7 and kb+16..kb+23, kb=8*(L>>4)
// B 32x16: lane L -> col base+(L&15), K = kb..kb+7 and kb+16..kb+23, kb=8*(L>>4)
// ---------------------------------------------------------------------------
__device__ __forceinline__ void split_frag_a(const float* Ls, int rowBase, int lane,
                                             v16bf* hi, v16bf* lo) {
  const float* p = Ls + (rowBase + (lane & 15)) * LDSW + ((lane >> 4) << 3);
#pragma unroll
  for (int i = 0; i < 8; ++i) {
    float f  = p[i];       bf16 h  = (bf16)f;
    (*hi)[i]     = h;      (*lo)[i]     = (bf16)(f - (float)h);
    float f2 = p[16 + i];  bf16 h2 = (bf16)f2;
    (*hi)[i + 8] = h2;     (*lo)[i + 8] = (bf16)(f2 - (float)h2);
  }
}

__device__ __forceinline__ void split_frag_b(const float* Ls, int rowBase, int lane,
                                             v16bf* hi, v16bf* lo) {
  const float* p = Ls + (rowBase + (lane & 15)) * LDSW + ((lane >> 4) << 3);
#pragma unroll
  for (int i = 0; i < 16; ++i) {
    float f = p[i + (i & 8)]; bf16 h = (bf16)f;
    (*hi)[i] = h;   (*lo)[i] = (bf16)(f - (float)h);
  }
}

__device__ __forceinline__ v8f wmma3(v16bf ah, v16bf al, v16bf bh, v16bf bl, v8f acc) {
  acc = __builtin_amdgcn_wmma_f32_16x16x32_bf16(false, ah, false, bh, (short)0, acc, false, false);
  acc = __builtin_amdgcn_wmma_f32_16x16x32_bf16(false, ah, false, bl, (short)0, acc, false, false);
  acc = __builtin_amdgcn_wmma_f32_16x16x32_bf16(false, al, false, bh, (short)0, acc, false, false);
  return acc;
}

__device__ __forceinline__ float silu_f(float v) { return v / (1.0f + expf(-v)); }

// ---------------------------------------------------------------------------
// Generic split-bf16 WMMA GEMM:  C[M,N] (+)= epi( A[M,K] * B[N,K]^T )
// grid = (N/64, M/128, nZ); per z: A += z*aColPerZ, B += z*bOffPerZ,
// C += z*cColPerZ  (block-diagonal support).
// epi: 0 = none, 1 = SiLU, 2 = gain(channel) * SiLU  (depth branch)
// TDM double-buffered staging: wave 0 issues DMAs, TENSORcnt-waited, one
// workgroup barrier per K chunk publishes the buffer and fences reuse.
// ---------------------------------------------------------------------------
__global__ __launch_bounds__(256) void k_gemm(
    const float* __restrict__ A, const float* __restrict__ B, float* __restrict__ C,
    int N, int K, int lda, int ldb, int ldc,
    int aColPerZ, int bOffPerZ, int cColPerZ,
    int epi, int accum,
    const float* __restrict__ logAd, const float* __restrict__ logdtd,
    const int* __restrict__ activeK)
{
  __shared__ float As[2][BM * LDSW];
  __shared__ float Bs[2][BN * LDSW];

  const int tid  = threadIdx.x;
  const int lane = tid & 31;
  const int wave = tid >> 5;
  const int wr   = (wave & 3) * 32;   // wave row offset within BM
  const int wc   = (wave >> 2) * 32;  // wave col offset within BN
  const int z    = blockIdx.z;

  A += (long)z * aColPerZ;
  B += (long)z * bOffPerZ;
  const int rowBase = blockIdx.y * BM;
  const int colBase = blockIdx.x * BN;
  if (colBase >= N) return;
  C += (long)z * cColPerZ;

  if (accum && tid < BM)   // warm the read-modify-write tile (global_prefetch_b8)
    __builtin_prefetch(&C[(long)(rowBase + tid) * ldc + colBase], 1, 3);

  const int nk = K / KC;
  if (wave == 0) {
    tdm_load_tile(A + (long)rowBase * lda, &As[0][0], BM, (unsigned)lda);
    tdm_load_tile(B + (long)colBase * ldb, &Bs[0][0], BN, (unsigned)ldb);
  }

  v8f acc[2][2];
#pragma unroll
  for (int i = 0; i < 2; ++i)
#pragma unroll
    for (int j = 0; j < 2; ++j)
#pragma unroll
      for (int v = 0; v < 8; ++v) acc[i][j][v] = 0.0f;

  for (int ki = 0; ki < nk; ++ki) {
    const int cur = ki & 1;
    if (wave == 0) __builtin_amdgcn_s_wait_tensorcnt(0);  // chunk ki landed
    __syncthreads();  // publish buffer `cur`; fences reuse of buffer cur^1
    if (wave == 0 && ki + 1 < nk) {                       // overlap next DMA
      tdm_load_tile(A + (long)rowBase * lda + (ki + 1) * KC, &As[cur ^ 1][0],
                    BM, (unsigned)lda);
      tdm_load_tile(B + (long)colBase * ldb + (ki + 1) * KC, &Bs[cur ^ 1][0],
                    BN, (unsigned)ldb);
    }

    v16bf a0h, a0l, a1h, a1l, b0h, b0l, b1h, b1l;
    split_frag_a(&As[cur][0], wr,      lane, &a0h, &a0l);
    split_frag_a(&As[cur][0], wr + 16, lane, &a1h, &a1l);
    split_frag_b(&Bs[cur][0], wc,      lane, &b0h, &b0l);
    split_frag_b(&Bs[cur][0], wc + 16, lane, &b1h, &b1l);

    acc[0][0] = wmma3(a0h, a0l, b0h, b0l, acc[0][0]);
    acc[0][1] = wmma3(a0h, a0l, b1h, b1l, acc[0][1]);
    acc[1][0] = wmma3(a1h, a1l, b0h, b0l, acc[1][0]);
    acc[1][1] = wmma3(a1h, a1l, b1h, b1l, acc[1][1]);
  }

  const float Kf = (epi == 2) ? (float)(*activeK) : 0.0f;

#pragma unroll
  for (int i = 0; i < 2; ++i) {
#pragma unroll
    for (int j = 0; j < 2; ++j) {
      int col  = colBase + wc + j * 16 + (lane & 15);
      int row0 = rowBase + wr + i * 16 + ((lane >> 4) << 3);
      float g = 1.0f;
      if (epi == 2) {
        int ch = z * cColPerZ + col;       // global channel within H
        if (ch < NMEM) {
          g = Kf;                          // a == 1 -> gain == K
        } else {
          float a  = expf(-expf(logAd[ch - NMEM]) * expf(logdtd[ch - NMEM]));
          float om = 1.0f - a;
          g = (fabsf(om) < 1e-6f) ? Kf : (1.0f - powf(a, Kf)) / fmaxf(om, 1e-8f);
        }
      }
#pragma unroll
      for (int v = 0; v < 8; ++v) {
        float val = acc[i][j][v];
        if (epi >= 1) val = silu_f(val);
        if (epi == 2) val *= g;
        long off = (long)(row0 + v) * ldc + col;
        if (accum) C[off] += val; else C[off] = val;
      }
    }
  }
}

// ---------------------------------------------------------------------------
// x[t,d] = wte[idx[t],d] + wpe[t,d]
// ---------------------------------------------------------------------------
__global__ void k_embed(const int* __restrict__ idx, const float* __restrict__ wte,
                        const float* __restrict__ wpe, float* __restrict__ x) {
  int i = blockIdx.x * 256 + threadIdx.x;
  int t = i >> 10, d = i & 1023;
  x[i] = wte[(long)idx[t] * DD + d] + wpe[i];
}

// a_seq[c] = max(exp(-exp(logA)*exp(logdt)), 1e-6);  apow[s,c] = a^(s+1)
__global__ void k_prep(const float* __restrict__ logA, const float* __restrict__ logdt,
                       float* __restrict__ aseq, float* __restrict__ apow) {
  int i = blockIdx.x * 256 + threadIdx.x;   // i < SLEN*HH
  int s = i >> 10, c = i & 1023;
  float a = fmaxf(expf(-expf(logA[c]) * expf(logdt[c])), 1e-6f);
  apow[i] = powf(a, (float)(s + 1));
  if (s == 0) aseq[c] = a;
}

// e = rmsnorm(x); cat[:,D:2D] = e; cat[:,2D:3D] = shift(e) (row 0 zeroed)
__global__ __launch_bounds__(256) void k_rms_src(const float* __restrict__ x,
                                                 float* __restrict__ cat) {
  int t = blockIdx.x, tid = threadIdx.x;
  __shared__ float red[256];
  float vals[4], s = 0.0f;
#pragma unroll
  for (int k = 0; k < 4; ++k) {
    float v = x[(long)t * DD + tid + k * 256];
    vals[k] = v; s += v * v;
  }
  red[tid] = s; __syncthreads();
  for (int o = 128; o > 0; o >>= 1) { if (tid < o) red[tid] += red[tid + o]; __syncthreads(); }
  float scale = rsqrtf(red[0] / (float)DD + 1.1920929e-07f);
#pragma unroll
  for (int k = 0; k < 4; ++k) {
    int d = tid + k * 256;
    float e = vals[k] * scale;
    cat[(long)t * CATW + DD + d] = e;
    if (t + 1 < TT) cat[(long)(t + 1) * CATW + 2 * DD + d] = e;
    if (t == 0)     cat[(long)2 * DD + d] = 0.0f;
  }
}

// Local scans: chunk j, channel c: y = a*y + d ; dseq overwritten in place.
__global__ void k_scan_local(float* __restrict__ dseq, const float* __restrict__ aseq,
                             float* __restrict__ carry) {
  int c = blockIdx.y * 256 + threadIdx.x;
  int j = blockIdx.x;
  float a = aseq[c], y = 0.0f;
  long base = (long)j * SLEN * HH + c;
  for (int s = 0; s < SLEN; ++s) {
    y = fmaf(a, y, dseq[base + (long)s * HH]);
    dseq[base + (long)s * HH] = y;
  }
  carry[j * HH + c] = y;
}

// Sequential combine of chunk carries -> exclusive combined prefix per chunk.
__global__ void k_scan_combine(float* __restrict__ carry, const float* __restrict__ apow) {
  int c = blockIdx.x * 256 + threadIdx.x;
  float a128 = apow[(SLEN - 1) * HH + c];
  float Cv = 0.0f;
  for (int j = 0; j < SCHUNK; ++j) {
    float tmp = carry[j * HH + c];
    carry[j * HH + c] = Cv;           // exclusive prefix for chunk j
    Cv = fmaf(a128, Cv, tmp);
  }
}

// h = local + a^(s+1)*carry_excl + gdep ; cat[:,0:D] = rmsnorm(h)
__global__ __launch_bounds__(256) void k_rms_h(
    const float* __restrict__ dseq, const float* __restrict__ gdep,
    const float* __restrict__ carry, const float* __restrict__ apow,
    float* __restrict__ cat) {
  int t = blockIdx.x, tid = threadIdx.x;
  int j = t >> 7, s2 = t & 127;
  __shared__ float hrow[DD];
  __shared__ float red[256];
  float s = 0.0f;
#pragma unroll
  for (int k = 0; k < 4; ++k) {
    int d = tid + k * 256;
    float hv = dseq[(long)t * HH + d] + apow[s2 * HH + d] * carry[j * HH + d]
             + gdep[(long)t * HH + d];
    hrow[d] = hv; s += hv * hv;
  }
  red[tid] = s; __syncthreads();
  for (int o = 128; o > 0; o >>= 1) { if (tid < o) red[tid] += red[tid + o]; __syncthreads(); }
  float scale = rsqrtf(red[0] / (float)HH + 1.1920929e-07f);
#pragma unroll
  for (int k = 0; k < 4; ++k) {
    int d = tid + k * 256;
    cat[(long)t * CATW + d] = hrow[d] * scale;
  }
}

// ---------------------------------------------------------------------------
extern "C" void kernel_launch(void* const* d_in, const int* in_sizes, int n_in,
                              void* d_out, int out_size, void* d_ws, size_t ws_size,
                              hipStream_t stream) {
  const int*   idx       = (const int*)  d_in[0];
  const int*   activeK   = (const int*)  d_in[1];
  const float* wte       = (const float*)d_in[2];
  const float* wpe       = (const float*)d_in[3];
  const float* bseq      = (const float*)d_in[4];
  const float* logA_seq  = (const float*)d_in[5];
  const float* logdt_seq = (const float*)d_in[6];
  const float* bdep      = (const float*)d_in[7];
  const float* logA_dep  = (const float*)d_in[8];
  const float* logdt_dep = (const float*)d_in[9];
  const float* wpost     = (const float*)d_in[10];
  const float* wlocal    = (const float*)d_in[11];
  const float* lowA      = (const float*)d_in[12];
  const float* lowB      = (const float*)d_in[13];
  const float* W         = (const float*)d_in[14];
  float*       out       = (float*)d_out;

  // Workspace carve-out (~57 MB, f32)
  float* p     = (float*)d_ws;
  float* x     = p; p += (size_t)TT * DD;
  float* cat   = p; p += (size_t)TT * CATW;
  float* dseq  = p; p += (size_t)TT * HH;
  float* gdep  = p; p += (size_t)TT * HH;
  float* rbuf  = p; p += (size_t)TT * HH;
  float* tmpLR = p; p += (size_t)TT * RNK;
  float* aseq  = p; p += (size_t)HH;
  float* apow  = p; p += (size_t)SLEN * HH;
  float* carry = p; p += (size_t)SCHUNK * HH;

  dim3 blk(256);
  k_embed<<<TT * DD / 256, blk, 0, stream>>>(idx, wte, wpe, x);

  for (int l = 0; l < NLAYER; ++l) {
    k_prep<<<SLEN * HH / 256, blk, 0, stream>>>(logA_seq + l * HH, logdt_seq + l * HH,
                                                aseq, apow);
    k_rms_src<<<TT, blk, 0, stream>>>(x, cat);

    // dseq = silu(blockdiag(e, bseq))
    k_gemm<<<dim3(1, TT / BM, NBLK), blk, 0, stream>>>(
        cat + DD, bseq + (size_t)l * NBLK * 64 * 64, dseq,
        64, 64, CATW, 64, HH, 64, 64 * 64, 64, 1, 0, nullptr, nullptr, nullptr);
    // gdep = gain * silu(blockdiag(e, bdep))
    k_gemm<<<dim3(1, TT / BM, NBLK), blk, 0, stream>>>(
        cat + DD, bdep + (size_t)l * NBLK * 64 * 64, gdep,
        64, 64, CATW, 64, HH, 64, 64 * 64, 64, 2, 0,
        logA_dep + l * NFORG, logdt_dep + l * NFORG, activeK);

    // causal scan over time (chunked recurrence, in place on dseq)
    k_scan_local<<<dim3(SCHUNK, HH / 256), blk, 0, stream>>>(dseq, aseq, carry);
    k_scan_combine<<<HH / 256, blk, 0, stream>>>(carry, apow);
    k_rms_h<<<TT, blk, 0, stream>>>(dseq, gdep, carry, apow, cat);

    // rbuf = silu(blockdiag(cat, wpost)), bi = 192
    k_gemm<<<dim3(1, TT / BM, NBLK), blk, 0, stream>>>(
        cat, wpost + (size_t)l * NBLK * 64 * 192, rbuf,
        64, 192, CATW, 192, HH, 192, 64 * 192, 64, 1, 0, nullptr, nullptr, nullptr);
    // x += blockdiag(rbuf, wlocal)
    k_gemm<<<dim3(1, TT / BM, NBLK), blk, 0, stream>>>(
        rbuf, wlocal + (size_t)l * NBLK * 64 * 64, x,
        64, 64, HH, 64, DD, 64, 64 * 64, 64, 0, 1, nullptr, nullptr, nullptr);
    // tmpLR = rbuf @ lowB^T
    k_gemm<<<dim3(1, TT / BM, 1), blk, 0, stream>>>(
        rbuf, lowB + (size_t)l * RNK * HH, tmpLR,
        RNK, HH, HH, HH, RNK, 0, 0, 0, 0, 0, nullptr, nullptr, nullptr);
    // x += tmpLR @ lowA^T
    k_gemm<<<dim3(DD / BN, TT / BM, 1), blk, 0, stream>>>(
        tmpLR, lowA + (size_t)l * DD * RNK, x,
        DD, RNK, RNK, RNK, DD, 0, 0, 0, 0, 1, nullptr, nullptr, nullptr);
  }

  // final rmsnorm + lm_head
  k_rms_src<<<TT, blk, 0, stream>>>(x, cat);
  k_gemm<<<dim3(VV / BN, TT / BM, 1), blk, 0, stream>>>(
      cat + DD, W, out,
      VV, DD, CATW, DD, VV, 0, 0, 0, 0, 0, nullptr, nullptr, nullptr);
}
